// CausalSA_42717744726164
// MI455X (gfx1250) — matrix operand order repaired and checked by reading, not compiled, not run on hardware
//
#include <hip/hip_runtime.h>
#include <hip/hip_bf16.h>

// ---------------------------------------------------------------------------
// Causal self-attention for MI455X (gfx1250), bf16 WMMA pipeline:
//   1) fp32 -> bf16 convert of x, Wqkv, Wo into workspace (4-wide)
//   2) qkv = x @ Wqkv            (v_wmma_f32_16x16x32_bf16, fp32 accum)
//   3) flash attention           (S^T = K.Q^T; K/V double-buffered into LDS
//                                 with global_load_async_to_lds_b128 once per
//                                 workgroup; online softmax per-lane)
//   4) out = y @ Wo              (WMMA, fp32 output)
// ---------------------------------------------------------------------------

typedef __attribute__((ext_vector_type(16))) __bf16 v16bf;
typedef __attribute__((ext_vector_type(8)))  __bf16 v8bf;
typedef __attribute__((ext_vector_type(4)))  __bf16 v4bf;
typedef __attribute__((ext_vector_type(4)))  float  v4f;
typedef __attribute__((ext_vector_type(8)))  float  v8f;

#define B_   2
#define T_   2048
#define D_   1024
#define H_   16
#define DH_  64
#define TD_  3072   // 3*D
#define MT_  4096   // B*T

// ---- fragment loaders (layouts per CDNA5 ISA §7.12.2) ----------

__device__ __forceinline__ v16bf cat8(v8bf g0, v8bf g1) {
  return __builtin_shufflevector(g0, g1, 0, 1, 2, 3, 4, 5, 6, 7,
                                 8, 9, 10, 11, 12, 13, 14, 15);
}

// A-matrix 16x32 bf16, row-major source with leading dim ld.
__device__ __forceinline__ v16bf load_a16(const __bf16* p, int ld) {
  const int lane = threadIdx.x & 31;
  const __bf16* r = p + (lane & 15) * ld + (lane >> 4) * 8;
  return cat8(*(const v8bf*)(r), *(const v8bf*)(r + 16));
}

// B-matrix 32x16 bf16, row-major K x N source with leading dim ld.
__device__ __forceinline__ v16bf load_b16(const __bf16* p, int ld) {
  const int lane = threadIdx.x & 31;
  const __bf16* r = p + (lane & 15) + (lane >> 4) * 8 * ld;
  v16bf b;
#pragma unroll
  for (int s = 0; s < 8; ++s) { b[s] = r[s * ld]; b[s + 8] = r[(16 + s) * ld]; }
  return b;
}

__device__ __forceinline__ v8f wmma_bf16(v16bf a, v16bf b, v8f c) {
  return __builtin_amdgcn_wmma_f32_16x16x32_bf16(
      false, a, false, b, (short)0, c, false, false);
}

// ---- CDNA5 async global -> LDS (ASYNCcnt-tracked) -------------------------

__device__ __forceinline__ unsigned lds_addr32(const void* p) {
  return (unsigned)(uintptr_t)p;       // low 32 bits of generic ptr = LDS off
}

__device__ __forceinline__ void async_ld_b128(const void* gaddr, unsigned ldsoff) {
  asm volatile("global_load_async_to_lds_b128 %0, %1, off"
               :: "v"(ldsoff), "v"(gaddr) : "memory");
}

__device__ __forceinline__ void wait_async0() {
  asm volatile("s_wait_asynccnt 0" ::: "memory");
}

// ---- fp32 -> bf16 convert (4-wide) ----------------------------------------

__global__ void cvt_f32_bf16_kernel(const float* __restrict__ in,
                                    __bf16* __restrict__ out, int n4) {
  int i = blockIdx.x * blockDim.x + threadIdx.x;
  if (i < n4) {
    v4f f = *(const v4f*)(in + 4 * i);
    v4bf b;
#pragma unroll
    for (int j = 0; j < 4; ++j) b[j] = (__bf16)f[j];
    *(v4bf*)(out + 4 * i) = b;
  }
}

// ---- GEMM: C[M,N] = A[M,K] * B[K,N], bf16 in, fp32 accum -------------------

template <bool OUT_BF16>
__global__ void __launch_bounds__(256)
gemm_bf16_kernel(const __bf16* __restrict__ A, const __bf16* __restrict__ Bm,
                 void* __restrict__ Cp, int M, int N, int K) {
  const int w    = threadIdx.x >> 5;
  const int wm   = w >> 2;
  const int wn   = w & 3;
  const int mb   = blockIdx.y * 128 + wm * 64;
  const int nb   = blockIdx.x * 256 + wn * 64;
  const int lane = threadIdx.x & 31;
  const int l16  = lane & 15;
  const int hi   = lane >> 4;

  v8f acc[4][4];
#pragma unroll
  for (int i = 0; i < 4; ++i)
#pragma unroll
    for (int j = 0; j < 4; ++j)
#pragma unroll
      for (int r = 0; r < 8; ++r) acc[i][j][r] = 0.0f;

  for (int kk = 0; kk < K; kk += 32) {
    if (kk + 32 < K) {
      __builtin_prefetch(A + mb * K + kk + 32, 0, 0);
      __builtin_prefetch(Bm + (kk + 32) * N + nb, 0, 0);
    }
    v16bf a[4], b[4];
#pragma unroll
    for (int i = 0; i < 4; ++i)
      a[i] = load_a16(A + (mb + 16 * i) * K + kk, K);
#pragma unroll
    for (int j = 0; j < 4; ++j)
      b[j] = load_b16(Bm + kk * N + nb + 16 * j, N);
#pragma unroll
    for (int i = 0; i < 4; ++i)
#pragma unroll
      for (int j = 0; j < 4; ++j)
        acc[i][j] = wmma_bf16(a[i], b[j], acc[i][j]);
  }

#pragma unroll
  for (int i = 0; i < 4; ++i)
#pragma unroll
    for (int j = 0; j < 4; ++j)
#pragma unroll
      for (int r = 0; r < 8; ++r) {
        int row = mb + 16 * i + r + hi * 8;
        int col = nb + 16 * j + l16;
        if (OUT_BF16)
          ((__bf16*)Cp)[row * N + col] = (__bf16)acc[i][j][r];
        else
          ((float*)Cp)[row * N + col] = acc[i][j][r];
      }
}

// ---- Flash attention -------------------------------------------------------
// Grid (T/64, B*H), block 128 = 4 waves; each wave owns 16 query rows.
// All waves loop in lockstep over 32-key blocks of the workgroup's causal
// range. K/V tiles (32x64 bf16 each) are staged ONCE per workgroup into LDS
// via async global->LDS b128 copies, double-buffered so block k0+32 streams
// while block k0 is computed. Per block:
//   S^T = K.Q^T  (K A-frags + hoisted Q^T B-frags), per-lane online softmax,
//   P bounce through LDS, O += P.V with row-major V B-frags from LDS.

__global__ void __launch_bounds__(128)
flash_attn_kernel(const __bf16* __restrict__ qkv, __bf16* __restrict__ outb) {
  __shared__ __align__(16) __bf16 kbuf[2][32 * 64];   // 2 x 4KB
  __shared__ __align__(16) __bf16 vbuf[2][32 * 64];   // 2 x 4KB
  __shared__ __align__(16) __bf16 pshare[4 * 16 * 32];

  const int tid  = threadIdx.x;
  const int w    = tid >> 5;
  const int lane = tid & 31;
  const int l16  = lane & 15;
  const int hi   = lane >> 4;

  const int qbase = blockIdx.x * 64 + w * 16;
  const int bb    = blockIdx.y >> 4;     // batch
  const int h     = blockIdx.y & 15;     // head

  const __bf16* base = qkv + (bb * T_) * TD_;
  const __bf16* qptr = base + h * DH_;
  const __bf16* kptr = base + D_ + h * DH_;
  const __bf16* vptr = base + 2 * D_ + h * DH_;

  const int kend = blockIdx.x * 64 + 64;   // exclusive max key for this block

  // stage K/V block [k0, k0+32) into buffer `buf` (4 async b128 per thread)
  auto stage = [&](int k0, int buf) {
#pragma unroll
    for (int c = 0; c < 2; ++c) {
      int chunk = tid + c * 128;           // 256 chunks of 16B per 4KB tile
      int row = chunk >> 3;                // 0..31
      int col = (chunk & 7) * 8;           // 0..56
      async_ld_b128(kptr + (k0 + row) * TD_ + col,
                    lds_addr32(&kbuf[buf][row * 64 + col]));
      async_ld_b128(vptr + (k0 + row) * TD_ + col,
                    lds_addr32(&vbuf[buf][row * 64 + col]));
    }
  };

  stage(0, 0);                             // prefetch first block

  // Hoisted Q^T B-fragments: lane%16 = query, elements = dh (table order).
  v16bf qb[2];
#pragma unroll
  for (int st = 0; st < 2; ++st) {
    const __bf16* qp = qptr + (qbase + l16) * TD_ + st * 32 + hi * 8;
    qb[st] = cat8(*(const v8bf*)(qp), *(const v8bf*)(qp + 16));
  }

  float m_s = -3.0e38f, l_s = 0.0f;        // per-lane (query = l16) stats
  v8f o[4];
#pragma unroll
  for (int j = 0; j < 4; ++j)
#pragma unroll
    for (int r = 0; r < 8; ++r) o[j][r] = 0.0f;

  __bf16* pw = pshare + w * (16 * 32);
  const int qmax = qbase + 15;
  int cb = 0;

  for (int k0 = 0; k0 < kend; k0 += 32, cb ^= 1) {
    wait_async0();          // this wave's staged loads for buf[cb] complete
    __syncthreads();        // all waves' loads complete; prior reads done

    if (k0 + 32 < kend) stage(k0 + 32, cb ^ 1);

    if (k0 <= qmax) {
      // load all fragments up front so LDS reads overlap the softmax math
      v16bf ak[2][2], vb[4];
#pragma unroll
      for (int n = 0; n < 2; ++n)
#pragma unroll
        for (int st = 0; st < 2; ++st)
          ak[n][st] = load_a16(&kbuf[cb][(n * 16) * 64 + st * 32], 64);
#pragma unroll
      for (int j = 0; j < 4; ++j)
        vb[j] = load_b16(&vbuf[cb][j * 16], 64);

      // ---- S^T tiles ----
      v8f st_[2];
#pragma unroll
      for (int n = 0; n < 2; ++n) {
#pragma unroll
        for (int r = 0; r < 8; ++r) st_[n][r] = 0.0f;
        st_[n] = wmma_bf16(ak[n][0], qb[0], st_[n]);
        st_[n] = wmma_bf16(ak[n][1], qb[1], st_[n]);
#pragma unroll
        for (int r = 0; r < 8; ++r) st_[n][r] *= 0.125f;   // 1/sqrt(64)
      }

      // ---- causal mask (key = VGPR slot, query = lane) ----
      if (k0 + 31 > qbase) {
        int query = qbase + l16;
#pragma unroll
        for (int n = 0; n < 2; ++n)
#pragma unroll
          for (int r = 0; r < 8; ++r) {
            int key = k0 + n * 16 + r + hi * 8;
            if (key > query) st_[n][r] = -3.0e38f;
          }
      }

      // ---- online softmax: per-lane + one xor-16 shuffle ----
      float mx = st_[0][0];
#pragma unroll
      for (int n = 0; n < 2; ++n)
#pragma unroll
        for (int r = 0; r < 8; ++r) mx = fmaxf(mx, st_[n][r]);
      mx = fmaxf(mx, __shfl_xor(mx, 16));

      float mn    = fmaxf(m_s, mx);
      float alpha = __expf(m_s - mn);
      m_s = mn;

      float rs = 0.0f;
      v8bf pb[2];
#pragma unroll
      for (int n = 0; n < 2; ++n)
#pragma unroll
        for (int r = 0; r < 8; ++r) {
          float e = __expf(st_[n][r] - mn);
          rs += e;
          pb[n][r] = (__bf16)e;
        }
      rs += __shfl_xor(rs, 16);
      l_s = l_s * alpha + rs;

      // stage P[query][key] (keys consecutive per lane)
#pragma unroll
      for (int n = 0; n < 2; ++n)
        *(v8bf*)(pw + l16 * 32 + n * 16 + hi * 8) = pb[n];

      // rescale O with alpha broadcast to row slots
#pragma unroll
      for (int r = 0; r < 8; ++r) {
        float ar = __shfl(alpha, r + hi * 8, 16);
#pragma unroll
        for (int j = 0; j < 4; ++j) o[j][r] *= ar;
      }

      // same-wave LDS RAW for the P bounce
      asm volatile("s_wait_dscnt 0" ::: "memory");

      v16bf pa = load_a16(pw, 32);
#pragma unroll
      for (int j = 0; j < 4; ++j)
        o[j] = wmma_bf16(pa, vb[j], o[j]);
    }
  }

  // ---- normalize and store y (bf16) as [B*T, D] ----
#pragma unroll
  for (int r = 0; r < 8; ++r) {
    float lr  = __shfl(l_s, r + hi * 8, 16);
    float inv = 1.0f / lr;
    int row = qbase + r + hi * 8;
#pragma unroll
    for (int j = 0; j < 4; ++j) {
      float val = o[j][r] * inv;
      outb[(bb * T_ + row) * D_ + h * DH_ + j * 16 + l16] = (__bf16)val;
    }
  }
}

// ---------------------------------------------------------------------------

extern "C" void kernel_launch(void* const* d_in, const int* in_sizes, int n_in,
                              void* d_out, int out_size, void* d_ws, size_t ws_size,
                              hipStream_t stream) {
  const float* x    = (const float*)d_in[0];   // [B,T,D]
  const float* Wqkv = (const float*)d_in[1];   // [D,3D]
  const float* Wo   = (const float*)d_in[2];   // [D,D]
  float* out = (float*)d_out;                  // [B,T,D]

  char* ws = (char*)d_ws;
  __bf16* xb    = (__bf16*)(ws);               //  8,388,608 B
  __bf16* wqkvb = (__bf16*)(ws + 8388608);     //  6,291,456 B
  __bf16* wob   = (__bf16*)(ws + 14680064);    //  2,097,152 B
  __bf16* qkvb  = (__bf16*)(ws + 16777216);    // 25,165,824 B
  __bf16* attnb = (__bf16*)(ws + 41943040);    //  8,388,608 B (total 48MB)

  const int nx = MT_ * D_;
  const int nq = D_ * TD_;
  const int no = D_ * D_;

  cvt_f32_bf16_kernel<<<nx / 4 / 256, 256, 0, stream>>>(x, xb, nx / 4);
  cvt_f32_bf16_kernel<<<nq / 4 / 256, 256, 0, stream>>>(Wqkv, wqkvb, nq / 4);
  cvt_f32_bf16_kernel<<<no / 4 / 256, 256, 0, stream>>>(Wo, wob, no / 4);

  gemm_bf16_kernel<true><<<dim3(TD_ / 256, MT_ / 128), 256, 0, stream>>>(
      xb, wqkvb, qkvb, MT_, TD_, D_);

  flash_attn_kernel<<<dim3(T_ / 64, B_ * H_), 128, 0, stream>>>(qkvb, attnb);

  gemm_bf16_kernel<false><<<dim3(D_ / 256, MT_ / 128), 256, 0, stream>>>(
      attnb, wob, out, MT_, D_, D_);
}
